// KNRM_89953795047578
// MI455X (gfx1250) — matrix operand order repaired and checked by reading, not compiled, not run on hardware
//
#include <hip/hip_runtime.h>
#include <hip/hip_bf16.h>

typedef __attribute__((ext_vector_type(16))) _Float16 v16h;
typedef __attribute__((ext_vector_type(2)))  _Float16 h2v;
typedef __attribute__((ext_vector_type(8)))  float    v8f;

#define LQ    30
#define LD    128
#define EMB   50
#define KPAD  64     // K padded 50 -> 64 (two 16x16x32 WMMA K-steps)
#define MPAD  32     // query rows padded 30 -> 32 (two 16-row M tiles)
#define NK    21

__global__ __launch_bounds__(256)
void knrm_kernel(const int* __restrict__ query,
                 const int* __restrict__ doc,
                 const float* __restrict__ emb,
                 const float* __restrict__ W1, const float* __restrict__ B1,
                 const float* __restrict__ W2, const float* __restrict__ B2,
                 const float* __restrict__ W3, const float* __restrict__ B3,
                 float* __restrict__ out)
{
    __shared__ _Float16 qh[MPAD * KPAD];   //  4 KB  normalized query, f16
    __shared__ _Float16 dh[LD   * KPAD];   // 16 KB  normalized doc,   f16
    __shared__ float    M [MPAD * LD];     // 16 KB  cosine match matrix
    __shared__ float    phi[NK];
    __shared__ float    hh1[10];
    __shared__ float    hh2[5];

    const int b    = blockIdx.x;
    const int tid  = threadIdx.x;
    const int lane = tid & 31;
    const int wid  = tid >> 5;

    // ---------------- Phase A: gather + L2 normalize + f16 convert ----------
    // 160 rows total (32 padded query rows + 128 doc rows), one thread each.
    if (tid < 160) {
        _Float16* dst;
        long v = -1;                       // -1 => zero pad row
        if (tid < MPAD) {
            dst = &qh[tid * KPAD];
            if (tid < LQ) v = query[(size_t)b * LQ + tid];
        } else {
            const int j = tid - MPAD;
            dst = &dh[j * KPAD];
            v = doc[(size_t)b * LD + j];
        }
        const float* row = (v >= 0) ? (emb + (size_t)v * EMB) : nullptr;
        float ss = 0.f;
        if (row) {
            #pragma unroll
            for (int k = 0; k < EMB; ++k) { float x = row[k]; ss += x * x; }
        }
        const float inv = 1.0f / fmaxf(sqrtf(ss), 1e-12f); // F.normalize eps
        #pragma unroll
        for (int k = 0; k < KPAD; ++k) {
            float x = (row && k < EMB) ? row[k] * inv : 0.f;
            dst[k] = (_Float16)x;
        }
    }
    if (tid < NK) phi[tid] = 0.f;
    __syncthreads();

    // ---------------- Phase B: M = q * d^T via v_wmma_f32_16x16x32_f16 ------
    // 16 output tiles (2 M-tiles x 8 N-tiles), wave-uniform distribution over
    // 8 waves -> EXEC all-1s as WMMA requires. K=64 -> 2 WMMA per tile.
    {
        const int half = lane >> 4;        // 0 | 1
        const int lr   = lane & 15;
        for (int t = wid; t < 16; t += 8) {
            const int mt   = t >> 3;       // M tile 0..1
            const int nt   = t & 7;        // N tile 0..7
            const int mrow = mt * 16 + lr; // A row (lanes 0-15 and 16-31 both M=lr)
            const int nrow = nt * 16 + lr; // doc row feeding B column
            v8f c = {};
            #pragma unroll
            for (int ks = 0; ks < 2; ++ks) {
                v16h a, bb;
                #pragma unroll
                for (int p = 0; p < 8; ++p) {
                    // A 16x32 f16 layout: VGPR p<4 -> K 0..7 (+8*half),
                    //                     p>=4    -> K 16..23 (+8*half)
                    const int ka = ks * 32 + ((p < 4) ? 0 : 16) + 8 * half + 2 * (p & 3);
                    h2v pa = *(const h2v*)&qh[mrow * KPAD + ka];
                    a[2 * p]     = pa[0];
                    a[2 * p + 1] = pa[1];
                    // B 32x16 f16 layout: same K order as A (p<4 -> K 0..7, p>=4 -> K 16..23, +8*half)
                    const int kb = ks * 32 + ((p < 4) ? 0 : 16) + 8 * half + 2 * (p & 3);
                    h2v pb = *(const h2v*)&dh[nrow * KPAD + kb];
                    bb[2 * p]     = pb[0];
                    bb[2 * p + 1] = pb[1];
                }
                c = __builtin_amdgcn_wmma_f32_16x16x32_f16(
                        /*neg_a=*/false, a, /*neg_b=*/false, bb,
                        /*c_mod=*/(short)0, c, /*reuse_a=*/false, /*reuse_b=*/false);
            }
            // C/D f32 16x16 layout: VGPR v -> row v (lanes 0-15) / v+8 (16-31)
            #pragma unroll
            for (int v = 0; v < 8; ++v)
                M[(mt * 16 + v + 8 * half) * LD + nt * 16 + lr] = c[v];
        }
    }
    __syncthreads();

    // ---------------- Phase C: RBF soft binning (the hot loop) --------------
    // 630 (query-row, kernel) pairs; each sums exp(-(M-mu)^2/(2s^2)) over the
    // 128 doc columns, then log1p -> atomic add into phi. Consecutive threads
    // share a query row -> LDS broadcast reads of M.
    for (int idx = tid; idx < LQ * NK; idx += 256) {
        const int qi = idx / NK;
        const int kn = idx % NK;
        float mu, i2s;
        if (kn < NK - 1) { mu = -0.95f + 0.1f * (float)kn; i2s = 50.0f;     } // 1/(2*0.1^2)
        else             { mu = 1.0f;                      i2s = 500000.0f; } // 1/(2*0.001^2)
        float s = 0.f;
        #pragma unroll 8
        for (int j = 0; j < LD; ++j) {
            const float d = M[qi * LD + j] - mu;
            s += __expf(-d * d * i2s);
        }
        atomicAdd(&phi[kn], __logf(1.0f + s));   // ds_add_f32
    }
    __syncthreads();

    // ---------------- Phase D: tiny MLP 21 -> 10 -> 5 -> 1 ------------------
    if (tid < 10) {
        float a = B1[tid];
        #pragma unroll
        for (int k = 0; k < NK; ++k) a += phi[k] * W1[k * 10 + tid];
        hh1[tid] = fmaxf(a, 0.f);
    }
    __syncthreads();
    if (tid < 5) {
        float a = B2[tid];
        #pragma unroll
        for (int k = 0; k < 10; ++k) a += hh1[k] * W2[k * 5 + tid];
        hh2[tid] = fmaxf(a, 0.f);
    }
    __syncthreads();
    if (tid == 0) {
        float a = B3[0];
        #pragma unroll
        for (int k = 0; k < 5; ++k) a += hh2[k] * W3[k];
        out[b] = a;
    }
}

extern "C" void kernel_launch(void* const* d_in, const int* in_sizes, int n_in,
                              void* d_out, int out_size, void* d_ws, size_t ws_size,
                              hipStream_t stream) {
    const int*   query = (const int*)d_in[0];
    const int*   doc   = (const int*)d_in[1];
    const float* emb   = (const float*)d_in[2];
    const float* W1    = (const float*)d_in[3];
    const float* B1    = (const float*)d_in[4];
    const float* W2    = (const float*)d_in[5];
    const float* B2    = (const float*)d_in[6];
    const float* W3    = (const float*)d_in[7];
    const float* B3    = (const float*)d_in[8];
    const int batch = in_sizes[0] / LQ;    // 1024
    knrm_kernel<<<batch, 256, 0, stream>>>(query, doc, emb, W1, B1, W2, B2, W3, B3,
                                           (float*)d_out);
}
